// LiquidLinear_33328946217763
// MI455X (gfx1250) — compile-verified, matrix operand order checked by reading, not run on hardware
//
#include <hip/hip_runtime.h>
#include <hip/hip_bf16.h>

// Problem dims
#define B_    64
#define S_    1024
#define DIN_  512
#define H_    512
#define DOUT_ 512

typedef __attribute__((ext_vector_type(16))) __bf16 v16bf;
typedef __attribute__((ext_vector_type(8)))  float  v8f;

union FragBF {
    v16bf v;
    __bf16 h[16];
    unsigned short u[16];
    uint4 q[2];
};

// Native f32 -> bf16 (compiler emits v_cvt_pk_bf16_f32 pairs, RNE)
__device__ __forceinline__ unsigned short f2bf(float f) {
    __bf16 b = (__bf16)f;
    return __builtin_bit_cast(unsigned short, b);
}
__device__ __forceinline__ float bf2f(unsigned short h) {
    return __uint_as_float(((unsigned int)h) << 16);
}

__device__ __forceinline__ void wait_asynccnt0() {
#if __has_builtin(__builtin_amdgcn_s_wait_asynccnt)
    __builtin_amdgcn_s_wait_asynccnt(0);
#else
    asm volatile("s_wait_asynccnt 0x0" ::: "memory");
#endif
}

// Async copy of 16 bytes from global to LDS (per lane), tracked by ASYNCcnt.
__device__ __forceinline__ void async_g2l_b128(unsigned lds_off, unsigned long long gaddr) {
    asm volatile("global_load_async_to_lds_b128 %0, %1, off"
                 :: "v"(lds_off), "v"(gaddr) : "memory");
}

// ---------------------------------------------------------------------------
// Phase 0: convert Wi / Wr / Wo (f32) -> bf16 in workspace (contiguous)
// ---------------------------------------------------------------------------
__global__ void cvt_weights(const float* __restrict__ Wi, const float* __restrict__ Wr,
                            const float* __restrict__ Wo, unsigned short* __restrict__ dst) {
    const int N = H_ * DIN_; // 262144 (all three weights are 512x512)
    int i = blockIdx.x * blockDim.x + threadIdx.x;
    if (i < N)            dst[i] = f2bf(Wi[i]);
    else if (i < 2 * N)   dst[i] = f2bf(Wr[i - N]);
    else if (i < 3 * N)   dst[i] = f2bf(Wo[i - 2 * N]);
}

// ---------------------------------------------------------------------------
// Phase 1: xin[t,b,h] = sum_d x[b,t,d] * Wi[h,d] + bi[h]   (bf16 out, [S,B,H])
// One wave = one 16x16 output tile; K-loop of WMMA 16x16x32 bf16.
// Output row index r = t*64 + b  (so [S,B,H] row offset is just r*H).
// ---------------------------------------------------------------------------
__global__ __launch_bounds__(256) void xin_gemm(
    const float* __restrict__ x, const unsigned short* __restrict__ Wi_bf,
    const float* __restrict__ bi, unsigned short* __restrict__ xin_bf) {
    const int lane = threadIdx.x & 31;
    const int wave = threadIdx.x >> 5;
    const int half = lane >> 4;
    const int m    = lane & 15;
    const int n0   = (blockIdx.y * 8 + wave) * 16;
    const int m0   = blockIdx.x * 16;

    const int r = m0 + m;                // output row (t*64 + b)
    const int t = r >> 6, b = r & 63;
    const float* arow = x + ((size_t)b * S_ + t) * DIN_;

    const int nn = n0 + m;               // this lane's B-row / C-column
    const unsigned short* brow = Wi_bf + (size_t)nn * DIN_;

    v8f acc = {};
    for (int kk = 0; kk < DIN_; kk += 32) {
        FragBF a, bb;
        const float4* p0 = (const float4*)(arow + kk + half * 8);
        const float4* p1 = (const float4*)(arow + kk + 16 + half * 8);
        float4 a0 = p0[0], a1 = p0[1], a2 = p1[0], a3 = p1[1];
        float av[16] = { a0.x, a0.y, a0.z, a0.w, a1.x, a1.y, a1.z, a1.w,
                         a2.x, a2.y, a2.z, a2.w, a3.x, a3.y, a3.z, a3.w };
        #pragma unroll
        for (int i = 0; i < 16; ++i) a.h[i] = (__bf16)av[i];   // native cvt
        bb.q[0] = *(const uint4*)(brow + kk + half * 8);
        bb.q[1] = *(const uint4*)(brow + kk + 16 + half * 8);
        acc = __builtin_amdgcn_wmma_f32_16x16x32_bf16(
            false, a.v, false, bb.v, (short)0, acc, false, false);
    }

    const float biv = bi[nn];
    #pragma unroll
    for (int j = 0; j < 8; ++j) {
        const int mm = m0 + j + 8 * half;     // C layout: row = j + 8*half
        xin_bf[(size_t)mm * H_ + nn] = f2bf(acc[j] + biv);
    }
}

// ---------------------------------------------------------------------------
// Phase 2: the scan. Batch rows are independent -> 4 workgroups of 16 rows,
// each runs the full 1024-step recurrence with only a workgroup barrier per
// step. h double-buffered in LDS (bf16); Wr streamed from L2; the per-step
// xin slice (16 KB) is prefetched with async global->LDS copies one step
// ahead, drained by s_wait_asynccnt + the per-step barrier.
// 32 waves: wave w owns output columns [w*16, w*16+16).
// ---------------------------------------------------------------------------
__global__ __launch_bounds__(1024) void liquid_scan(
    const unsigned short* __restrict__ xin_bf, const unsigned short* __restrict__ Wr_bf,
    const float* __restrict__ br, const float* __restrict__ tau,
    unsigned short* __restrict__ hs_bf, float* __restrict__ hfinal) {
    __shared__ unsigned short hbuf[2][16][520];     // +8 pad vs bank conflicts (~33 KB)
    __shared__ unsigned short xstage[2][16 * 512];  // async-staged xin slices (32 KB)

    const int tid  = threadIdx.x;
    const int lane = tid & 31;
    const int wave = tid >> 5;
    const int half = lane >> 4;
    const int m    = lane & 15;
    const int n    = wave * 16 + m;      // this lane's column in H
    const int b0   = blockIdx.x * 16;    // batch-row block

    const float brv = br[n];
    const float dv  = 1.0f / (1.0f + __expf(-tau[n]));   // sigmoid(tau), loop-invariant
    const unsigned short* wrow = Wr_bf + (size_t)n * H_;

    // LDS byte offset of the staging buffers (low 32 bits of generic address)
    const unsigned xs_lds   = (unsigned)(size_t)(&xstage[0][0]);
    const unsigned long long xin_ga = (unsigned long long)(size_t)xin_bf;
    // xin slice for step t starts at byte offset ((t*64 + b0) * 512) * 2
    const unsigned long long slice0 = ((unsigned long long)b0) * 1024ull;

    // Kick off async stage of step 0 while zeroing h0.
    async_g2l_b128(xs_lds + (unsigned)(tid * 16), xin_ga + slice0 + (unsigned)(tid * 16));
    for (int i = tid; i < 2 * 16 * 520; i += 1024) (&hbuf[0][0][0])[i] = 0;
    wait_asynccnt0();
    __syncthreads();

    for (int t = 0; t < S_; ++t) {
        const int cur = t & 1, nxt = cur ^ 1;

        // Prefetch next step's xin slice into the other stage buffer.
        if (t + 1 < S_) {
            const unsigned long long gsl =
                xin_ga + (((unsigned long long)(t + 1) * 64ull + b0) * 1024ull);
            async_g2l_b128(xs_lds + (unsigned)(nxt * 16384 + tid * 16),
                           gsl + (unsigned)(tid * 16));
        }

        const unsigned short* hrow = &hbuf[cur][m][0];
        v8f acc = {};
        #pragma unroll 4
        for (int kk = 0; kk < H_; kk += 32) {
            FragBF a, bb;
            a.q[0]  = *(const uint4*)(hrow + kk + half * 8);        // ds_load_b128
            a.q[1]  = *(const uint4*)(hrow + kk + 16 + half * 8);
            bb.q[0] = *(const uint4*)(wrow + kk + half * 8);        // L2-resident Wr
            bb.q[1] = *(const uint4*)(wrow + kk + 16 + half * 8);
            acc = __builtin_amdgcn_wmma_f32_16x16x32_bf16(
                false, a.v, false, bb.v, (short)0, acc, false, false);
        }

        #pragma unroll
        for (int j = 0; j < 8; ++j) {
            const int mj = j + 8 * half;                    // C row within tile
            const size_t row = (size_t)t * B_ + (b0 + mj);  // [S,B] flat row
            const float xv   = bf2f(xstage[cur][mj * 512 + n]);
            const float I    = acc[j] + xv + brv;
            const float s    = tanhf(I);
            const float hold = bf2f(hbuf[cur][mj][n]);
            const float hn   = hold + dv * (s - hold);      // (1-d)*h + d*s
            const unsigned short hb = f2bf(hn);
            hbuf[nxt][mj][n]    = hb;
            hs_bf[row * H_ + n] = hb;
        }
        wait_asynccnt0();   // own async issues complete...
        __syncthreads();    // ...and all waves' stages visible / h swap safe
    }

    // After t=1023 the final h lives in buffer 0 (1024 swaps from buffer 0).
    #pragma unroll
    for (int j = 0; j < 8; ++j) {
        const int mj = j + 8 * half;
        hfinal[(size_t)(b0 + mj) * H_ + n] = bf2f(hbuf[0][mj][n]);
    }
}

// ---------------------------------------------------------------------------
// Phase 3: outs[b,t,o] = sum_h hs[t,b,h] * Wo[o,h] + bo[o]   (f32 out)
// ---------------------------------------------------------------------------
__global__ __launch_bounds__(256) void out_gemm(
    const unsigned short* __restrict__ hs_bf, const unsigned short* __restrict__ Wo_bf,
    const float* __restrict__ bo, float* __restrict__ out) {
    const int lane = threadIdx.x & 31;
    const int wave = threadIdx.x >> 5;
    const int half = lane >> 4;
    const int m    = lane & 15;
    const int n0   = (blockIdx.y * 8 + wave) * 16;
    const int m0   = blockIdx.x * 16;

    const unsigned short* arow = hs_bf + (size_t)(m0 + m) * H_;  // row r = t*64+b
    const int nn = n0 + m;
    const unsigned short* brow = Wo_bf + (size_t)nn * H_;

    v8f acc = {};
    for (int kk = 0; kk < H_; kk += 32) {
        FragBF a, bb;
        a.q[0]  = *(const uint4*)(arow + kk + half * 8);
        a.q[1]  = *(const uint4*)(arow + kk + 16 + half * 8);
        bb.q[0] = *(const uint4*)(brow + kk + half * 8);
        bb.q[1] = *(const uint4*)(brow + kk + 16 + half * 8);
        acc = __builtin_amdgcn_wmma_f32_16x16x32_bf16(
            false, a.v, false, bb.v, (short)0, acc, false, false);
    }

    const float bov = bo[nn];
    #pragma unroll
    for (int j = 0; j < 8; ++j) {
        const int mm = m0 + j + 8 * half;
        const int t = mm >> 6, b = mm & 63;
        out[(size_t)b * ((size_t)S_ * DOUT_) + (size_t)t * DOUT_ + nn] = acc[j] + bov;
    }
}

// ---------------------------------------------------------------------------
extern "C" void kernel_launch(void* const* d_in, const int* in_sizes, int n_in,
                              void* d_out, int out_size, void* d_ws, size_t ws_size,
                              hipStream_t stream) {
    const float* x   = (const float*)d_in[0];
    const float* Wi  = (const float*)d_in[1];
    const float* bi  = (const float*)d_in[2];
    const float* Wr  = (const float*)d_in[3];
    const float* br  = (const float*)d_in[4];
    const float* tau = (const float*)d_in[5];
    const float* Wo  = (const float*)d_in[6];
    const float* bo  = (const float*)d_in[7];

    float* out    = (float*)d_out;                       // [B,S,DOUT]
    float* hfinal = out + (size_t)B_ * S_ * DOUT_;       // [B,H] appended

    // Workspace layout (bf16 everywhere): weights | xin[S,B,H] | hs[S,B,H]
    unsigned short* wts    = (unsigned short*)d_ws;
    const size_t WN        = (size_t)H_ * DIN_;          // 262144
    unsigned short* Wi_bf  = wts;
    unsigned short* Wr_bf  = wts + WN;
    unsigned short* Wo_bf  = wts + 2 * WN;
    unsigned short* xin_bf = wts + 3 * WN;
    unsigned short* hs_bf  = xin_bf + (size_t)S_ * B_ * H_;

    cvt_weights<<<(3 * (int)WN + 255) / 256, 256, 0, stream>>>(Wi, Wr, Wo, wts);
    xin_gemm<<<dim3((S_ * B_) / 16, H_ / (16 * 8)), 256, 0, stream>>>(x, Wi_bf, bi, xin_bf);
    liquid_scan<<<B_ / 16, 1024, 0, stream>>>(xin_bf, Wr_bf, br, tau, hs_bf, hfinal);
    out_gemm<<<dim3((S_ * B_) / 16, DOUT_ / (16 * 8)), 256, 0, stream>>>(hs_bf, Wo_bf, bo, out);
}
